// BinaryMaskPredictor_47193100649104
// MI455X (gfx1250) — matrix operand order repaired and checked by reading, not compiled, not run on hardware
//
#include <hip/hip_runtime.h>

// ---------------------------------------------------------------------------
// MI455X (gfx1250). conv1 as implicit GEMM (M=64oc x N=1024px x K=1152 per
// anchor/oc-block) on v_wmma_f32_16x16x32_bf16. K reordered to
// k' = (kh*3+kw)*128 + ic so A and B fragments are contiguous 16B LDS runs
// (ds_load_b128), with padded strides for conflict-free banking.
// K2 stages h tiles via the Tensor Data Mover (iterated 2D tile with LDS
// padding) when the builtin exists.
// ---------------------------------------------------------------------------

typedef __attribute__((ext_vector_type(16))) __bf16 v16bf;
typedef __attribute__((ext_vector_type(8)))  float  v8f;
typedef __attribute__((ext_vector_type(4)))  unsigned u32x4;
typedef __attribute__((ext_vector_type(8)))  int      i32x8;
typedef __attribute__((ext_vector_type(4)))  int      i32x4;

__device__ __forceinline__ unsigned short f2bf(float f) {
  unsigned u = __float_as_uint(f);
  unsigned r = u + 0x7FFFu + ((u >> 16) & 1u);   // round-to-nearest-even
  return (unsigned short)(r >> 16);
}
__device__ __forceinline__ float bf2f(unsigned short h) {
  return __uint_as_float(((unsigned)h) << 16);
}

// LDS geometry for K1 (bytes)
#define SA_STRIDE   1160                     // padded A row stride (elems)
#define SA_BYTES    (64 * SA_STRIDE * 2)     // 148480
#define SIN_CSTR    136                      // padded ic stride (elems)
#define SIN_BYTES   (10 * 34 * SIN_CSTR * 2) // 92480
#define SB1_OFF     (SA_BYTES + SIN_BYTES)   // 240960
#define SMEM1_BYTES (SB1_OFF + 256)          // 241216

// ---------------------------------------------------------------------------
// K0: W1 (256,128,3,3) f32 -> bf16 with K reorder: w1b[oc][q*128+ic],
// q = kh*3+kw.
// ---------------------------------------------------------------------------
__global__ void k_prep_w1(const float* __restrict__ W1,
                          unsigned short* __restrict__ w1b) {
  int i = blockIdx.x * 256 + threadIdx.x;          // destination index
  if (i < 294912) {
    int oc  = i / 1152;
    int rem = i - oc * 1152;
    int q   = rem >> 7;
    int ic  = rem & 127;
    w1b[i] = f2bf(W1[oc * 1152 + ic * 9 + q]);
  }
}

// ---------------------------------------------------------------------------
// K1: conv1 + bias + ReLU via WMMA bf16.
// grid = (256 anchors, 4 oc-blocks of 64), block = 256 threads (8 waves).
// ---------------------------------------------------------------------------
__global__ __launch_bounds__(256) void k_conv1_wmma(
    const float* __restrict__ fmap,      // (128,320,320)
    const int*   __restrict__ anchors,   // (256,4) = x0,x1,y0,y1
    const float* __restrict__ b1,        // (256)
    const unsigned short* __restrict__ w1b,   // (256,1152) bf16, K-reordered
    unsigned short* __restrict__ hbuf)        // (256,256,32,32) bf16 out
{
  extern __shared__ char smem[];
  unsigned short* sA  = (unsigned short*)smem;               // 64 x 1160
  unsigned short* sIn = (unsigned short*)(smem + SA_BYTES);  // 10*34 x 136
  float*          sB1 = (float*)(smem + SB1_OFF);            // 64

  const int n   = blockIdx.x;
  const int ocb = blockIdx.y;
  const int tid = threadIdx.x;
  const int x0  = anchors[n * 4 + 0];
  const int y0  = anchors[n * 4 + 2];

  // Stage A-block with padded row stride, as dwords.
  {
    const unsigned* src = (const unsigned*)(w1b + ocb * 64 * 1152);
    unsigned*       dst = (unsigned*)sA;
    for (int i = tid; i < 36864; i += 256) {     // 64 rows x 576 dw
      int m = i / 576, kd = i - m * 576;
      dst[m * 580 + kd] = src[i];
    }
  }
  if (tid < 64) sB1[tid] = b1[ocb * 64 + tid];

  const int w    = tid >> 5;       // wave id (wave32)
  const int lane = tid & 31;
  const int mt   = w >> 1;         // M-tile 0..3 (16 oc each)
  const int half = w & 1;          // which half of px tiles
  const int lm   = lane & 15;
  const int lh   = lane >> 4;
  const int aRow = (mt * 16 + lm) * SA_STRIDE;

  for (int g = 0; g < 4; ++g) {    // 4 strips of 8 output rows
    __syncthreads();               // prior strip fully consumed
    // Stage input strip rows [8g-1, 8g+8], cols [-1,32], layout [r][c][ic].
    for (int i = tid; i < 10 * 128 * 34; i += 256) {
      int c  = i % 34;
      int t  = i / 34;
      int ic = t & 127;
      int r  = t >> 7;
      int cy = 8 * g - 1 + r;      // crop y
      int cx = c - 1;              // crop x
      float v = 0.f;
      if ((unsigned)cy < 32u && (unsigned)cx < 32u)
        v = fmap[ic * 102400 + (y0 + cy) * 320 + (x0 + cx)];
      sIn[(r * 34 + c) * SIN_CSTR + ic] = f2bf(v);
    }
    __syncthreads();

    // Prefetch next strip's rows into cache while we compute.
    if (g < 3) {
      int ic = tid & 127;
      int cy = 8 * (g + 1) - 1 + (tid >> 7) * 9;
      if ((unsigned)cy < 32u)
        __builtin_prefetch(&fmap[ic * 102400 + (y0 + cy) * 320 + x0], 0, 1);
    }

    for (int tt = 0; tt < 8; ++tt) {           // 8 px tiles per wave
      const int t   = half * 8 + tt;           // tile in strip (16 total)
      const int rl  = t >> 1;                  // local row 0..7
      const int cx0 = (t & 1) * 16;            // col base 0 or 16
      const int px  = cx0 + lm;                // this lane's pixel col (N)
      v8f acc = {};
      for (int kk = 0; kk < 36; ++kk) {
        const int kbase = kk * 32;
        // A fragment: lane row M=lm; k-run [kbase+16lh, +16).
        union { v16bf v; u32x4 q[2]; } A;
        const unsigned short* ap = sA + aRow + kbase + 16 * lh;
        A.q[0] = *(const u32x4*)(ap);
        A.q[1] = *(const u32x4*)(ap + 8);
        // B fragment: lane col N=px; elements j -> k' = kbase + 16*lh + j,
        // i.e. 16 consecutive ic at one spatial tap -> contiguous 32B.
        union { v16bf v; u32x4 q[2]; } B;
        const int k0  = kbase + lh * 16;
        const int qq  = k0 >> 7;               // kh*3+kw
        const int icb = k0 & 127;
        const int kh  = qq / 3;
        const int kw  = qq - kh * 3;
        const unsigned short* bp =
            sIn + ((rl + kh) * 34 + px + kw) * SIN_CSTR + icb;
        B.q[0] = *(const u32x4*)(bp);
        B.q[1] = *(const u32x4*)(bp + 8);
        acc = __builtin_amdgcn_wmma_f32_16x16x32_bf16(
            false, A.v, false, B.v, (short)0, acc, false, false);
      }
      // Epilogue: D layout -> M = i + 8*lh, N = lm. bias + ReLU + bf16 store.
      const int py = g * 8 + rl;
      #pragma unroll
      for (int i = 0; i < 8; ++i) {
        const int mo = mt * 16 + i + 8 * lh;          // oc within block
        const int oc = ocb * 64 + mo;
        float v = fmaxf(acc[i] + sB1[mo], 0.f);
        hbuf[((n * 256 + oc) * 32 + py) * 32 + px] = f2bf(v);
      }
    }
  }
}

// ---------------------------------------------------------------------------
// K2: conv2 (256->1, 3x3 SAME) + bias + BCE-with-logits, per-anchor sum.
// h staged to LDS in 16-channel chunks with zero halo; staging done by the
// Tensor Data Mover when available (one iterated 2D-tile descriptor covers
// all 16 channels, LDS padding 1dw/16dw produces the 34-col stride).
// ---------------------------------------------------------------------------
__global__ __launch_bounds__(256) void k_conv2_bce(
    const unsigned short* __restrict__ hbuf,
    const float* __restrict__ W2,    // (1,256,3,3)
    const float* __restrict__ b2,    // (1)
    const int*   __restrict__ seg,   // (1280,1280)
    const int*   __restrict__ anchors,
    const int*   __restrict__ labels,
    const int*   __restrict__ base_classes,
    float*       __restrict__ per_anchor)
{
  __shared__ unsigned short sh[16 * 34 * 34];   // 16 ch, zero-padded 34x34
  __shared__ float sW2[2304];
  __shared__ float sred[256];

  const int n   = blockIdx.x;
  const int tid = threadIdx.x;
  for (int i = tid; i < 2304; i += 256) sW2[i] = W2[i];

  float acc[4] = {0.f, 0.f, 0.f, 0.f};
  for (int occ = 0; occ < 16; ++occ) {
    __syncthreads();                             // prior chunk consumed
    for (int i = tid; i < 16 * 34 * 34; i += 256) sh[i] = 0;
    __syncthreads();
#if __has_builtin(__builtin_amdgcn_tensor_load_to_lds)
    if (tid < 32) {  // wave 0 issues one TDM op covering all 16 channels
      unsigned lds_base =
          (unsigned)(uintptr_t)&sh[35];          // interior (y=0,x=0) cell
      unsigned long long ga = (unsigned long long)(uintptr_t)(
          hbuf + (size_t)(n * 256 + occ * 16) * 1024);
      u32x4 g0;
      g0.x = 1u;                                  // count=1 (valid), user mode
      g0.y = lds_base;                            // lds_addr
      g0.z = (unsigned)(ga & 0xFFFFFFFFu);        // global_addr[31:0]
      g0.w = (unsigned)((ga >> 32) & 0x1FFFFFFu)  // global_addr[56:32]
             | (2u << 30);                        // type = 2 ("image")
      i32x8 g1;
      g1[0] = (int)((1u << 16)        // data_size = 2B
                  | (1u << 19)        // iterate_enable
                  | (1u << 20)        // pad_enable
                  | (3u << 22));      // pad_interval = 16 dw, pad_amount = 1 dw
      g1[1] = (int)(32u << 16);       // tensor_dim0 = 32 (bits [79:48] lo)
      g1[2] = (int)(32u << 16);       // tensor_dim0 hi=0 | tensor_dim1 lo = 32
      g1[3] = (int)(32u << 16);       // tensor_dim1 hi=0 | tile_dim0 = 32
      g1[4] = 32;                     // tile_dim1 = 32, tile_dim2 = 0
      g1[5] = 32;                     // tensor_dim0_stride = 32
      g1[6] = 0;                      // stride hi | tensor_dim1_stride lo
      g1[7] = 0;
      i32x4 g2;
      g2[0] = 0;                      // tensor_dim2 (unused w/ iterate)
      g2[1] = 1156;                   // lds_addr_increment (data units)
      g2[2] = 1024;                   // global_addr_increment (data units)
      g2[3] = (15 << 16);             // iterate_count = 15 -> 16 iterations
      i32x4 g3 = {0, 0, 0, 0};
      i32x8 g4 = {0, 0, 0, 0, 0, 0, 0, 0};       // extra operand (clang-23 form)
      __builtin_amdgcn_tensor_load_to_lds(g0, g1, g2, g3, g4, 0);
      __builtin_amdgcn_s_wait_tensorcnt(0);
    }
#else
    for (int i = tid; i < 16 * 1024; i += 256) {
      const int oc = i >> 10, p = i & 1023, y = p >> 5, x = p & 31;
      sh[oc * 1156 + (y + 1) * 34 + (x + 1)] =
          hbuf[((n * 256 + occ * 16 + oc) * 32 + y) * 32 + x];
    }
#endif
    __syncthreads();
    #pragma unroll
    for (int pi = 0; pi < 4; ++pi) {
      const int p = tid + pi * 256, py = p >> 5, px = p & 31;
      float a = acc[pi];
      for (int oc = 0; oc < 16; ++oc) {
        const float* wq = &sW2[(occ * 16 + oc) * 9];
        const unsigned short* hb = &sh[oc * 1156 + py * 34 + px];
        #pragma unroll
        for (int kh = 0; kh < 3; ++kh)
          #pragma unroll
          for (int kw = 0; kw < 3; ++kw)
            a = fmaf(wq[kh * 3 + kw], bf2f(hb[kh * 34 + kw]), a);
      }
      acc[pi] = a;
    }
  }

  const int x0 = anchors[n * 4 + 0], y0 = anchors[n * 4 + 2];
  const int cls = base_classes[labels[n]];
  const float bias2 = b2[0];
  float s = 0.f;
  #pragma unroll
  for (int pi = 0; pi < 4; ++pi) {
    const int p = tid + pi * 256, py = p >> 5, px = p & 31;
    const float l = acc[pi] + bias2;
    const int sv = seg[((y0 + py) * 4) * 1280 + (x0 + px) * 4];
    const float tgt = (sv == cls) ? 1.f : 0.f;
    s += fmaxf(l, 0.f) - l * tgt + log1pf(expf(-fabsf(l)));
  }
  sred[tid] = s;
  __syncthreads();
  for (int off = 128; off > 0; off >>= 1) {
    if (tid < off) sred[tid] += sred[tid + off];
    __syncthreads();
  }
  if (tid == 0) per_anchor[n] = sred[0];
}

// ---------------------------------------------------------------------------
// K3: deterministic final reduction (replay-safe, no atomics).
// ---------------------------------------------------------------------------
__global__ void k_final(const float* __restrict__ per_anchor,
                        float* __restrict__ out) {
  __shared__ float sred[256];
  const int tid = threadIdx.x;
  sred[tid] = per_anchor[tid];
  __syncthreads();
  for (int off = 128; off > 0; off >>= 1) {
    if (tid < off) sred[tid] += sred[tid + off];
    __syncthreads();
  }
  if (tid == 0) out[0] = (sred[0] * (1.0f / 1024.0f)) / (256.0f + 1e-10f);
}

// ---------------------------------------------------------------------------
extern "C" void kernel_launch(void* const* d_in, const int* in_sizes, int n_in,
                              void* d_out, int out_size, void* d_ws,
                              size_t ws_size, hipStream_t stream) {
  (void)in_sizes; (void)n_in; (void)out_size; (void)ws_size;
  const float* fmap    = (const float*)d_in[0];
  const int*   seg     = (const int*)d_in[1];
  const int*   anchors = (const int*)d_in[2];
  const int*   labels  = (const int*)d_in[3];
  const int*   basec   = (const int*)d_in[4];
  const float* W1      = (const float*)d_in[5];
  const float* b1      = (const float*)d_in[6];
  const float* W2      = (const float*)d_in[7];
  const float* b2      = (const float*)d_in[8];

  char* ws = (char*)d_ws;
  unsigned short* w1b        = (unsigned short*)(ws);              // 576 KB
  float*          per_anchor = (float*)(ws + 655360);              // 1 KB
  unsigned short* hbuf       = (unsigned short*)(ws + (1u << 20)); // 128 MB

  k_prep_w1<<<dim3((294912 + 255) / 256), dim3(256), 0, stream>>>(W1, w1b);

  (void)hipFuncSetAttribute((const void*)k_conv1_wmma,
                            hipFuncAttributeMaxDynamicSharedMemorySize,
                            (int)SMEM1_BYTES);
  k_conv1_wmma<<<dim3(256, 4), dim3(256), SMEM1_BYTES, stream>>>(
      fmap, anchors, b1, w1b, hbuf);

  k_conv2_bce<<<dim3(256), dim3(256), 0, stream>>>(
      hbuf, W2, b2, seg, anchors, labels, basec, per_anchor);

  k_final<<<dim3(1), dim3(256), 0, stream>>>(per_anchor, (float*)d_out);
}
